// BlockSparseCausalConv_17497696763963
// MI455X (gfx1250) — matrix operand order repaired and checked by reading, not compiled, not run on hardware
//
#include <hip/hip_runtime.h>

// ---------------- problem constants (from reference setup) ----------------
#define B_    4
#define C_    2048
#define T_    2048
#define NBLK  128
#define BS    64
#define KS    16
#define NBR   32           // C_/BS
#define TP    16           // zero-pad rows in time (>= KS) for causal left edge
#define NT    128          // T columns per workgroup

typedef __attribute__((ext_vector_type(16))) _Float16 v16h;
typedef __attribute__((ext_vector_type(8)))  float    v8f;

// ---------------- workspace layout ----------------
// Wf: weights pre-swizzled into exact WMMA A-fragment order, f16.
//     [blk][mt(4)][kc(32)][lane(32)][e(16)]  -> 8,388,608 halfs = 16 MiB
// Xt: x transposed+converted: [b][TP + t][c], f16, c fastest.
//     4 * 2064 * 2048 halfs = 33.8 MB.  Total ws use ~51 MB (L2-resident).
#define WF_ELEMS  ((size_t)NBLK * 4 * 32 * 32 * 16)
#define WF_BYTES  (WF_ELEMS * 2)
#define XT_ROWS   (T_ + TP)

// ---------------- prep: zero output (harness poisons it) ----------------
__global__ void zero_f32(float* __restrict__ p, size_t n) {
    size_t i = (size_t)blockIdx.x * blockDim.x + threadIdx.x;
    size_t stride = (size_t)gridDim.x * blockDim.x;
    for (; i < n; i += stride) p[i] = 0.0f;
}

// ---------------- prep: swizzle weights into A-fragment order ----------------
// Global K index chosen as K = k*64 + i  (tap outer, channel inner).
// A layout (16x32 MxK), matched to B: lane = g*16 + m%16 (g = lane/16),
//   element e: Kloc = g*16 + e
__global__ void prep_w(const float* __restrict__ bv, _Float16* __restrict__ Wf) {
    size_t idx = (size_t)blockIdx.x * blockDim.x + threadIdx.x;
    if (idx >= WF_ELEMS) return;
    int e    = (int)(idx         & 15);
    int lane = (int)((idx >> 4)  & 31);
    int kc   = (int)((idx >> 9)  & 31);
    int mt   = (int)((idx >> 14) & 3);
    int blk  = (int)(idx >> 16);
    int g = lane >> 4;
    int m = mt * 16 + (lane & 15);
    int Kloc = g * 16 + e;
    int K = kc * 32 + Kloc;
    int k = K >> 6;        // tap
    int i = K & 63;        // input channel within block
    // block_values: (NBLK, BS_out, BS_in, KS)
    float w = bv[(((size_t)blk * BS + m) * BS + i) * KS + k];
    Wf[idx] = (_Float16)w;
}

// ---------------- prep: transpose + convert x to [b][TP+t][c] f16 ----------------
__global__ void prep_x(const float* __restrict__ x, _Float16* __restrict__ Xt) {
    __shared__ float tile[32][33];
    int t0 = blockIdx.x * 32, c0 = blockIdx.y * 32, b = blockIdx.z;
    int tx = threadIdx.x, ty = threadIdx.y;           // 32 x 8
    #pragma unroll
    for (int i = ty; i < 32; i += 8)
        tile[i][tx] = x[((size_t)b * C_ + c0 + i) * T_ + t0 + tx];
    __syncthreads();
    #pragma unroll
    for (int i = ty; i < 32; i += 8)
        Xt[((size_t)b * XT_ROWS + TP + t0 + i) * C_ + c0 + tx] = (_Float16)tile[tx][i];
}

__global__ void zero_xpad(_Float16* __restrict__ Xt) {
    int idx = blockIdx.x * blockDim.x + threadIdx.x;   // B_*TP*C_ threads
    int b = idx / (TP * C_);
    int rem = idx % (TP * C_);
    Xt[(size_t)b * XT_ROWS * C_ + rem] = (_Float16)0.0f;
}

// ---------------- main: WMMA implicit-im2col GEMM ----------------
// One workgroup: (batch b, block blk, 128-column T tile).
// 8 waves = 4 M-tiles x 2 N-halves; each wave: 16(M) x 64(N) via 4 accumulators.
// B-fragment (f16 32x16 KxN): lane = g*16 + n%16, element e -> Kloc = g*16 + e.
// With K = k*64 + i: chunk kc -> k = kc>>1 (constant per chunk),
// channel = col*64 + (kc&1)*32 + g*16 + e  => 16 contiguous halfs in Xt. Aligned 32B.
__global__ __launch_bounds__(256)
void bsconv_wmma(const _Float16* __restrict__ Wf, const _Float16* __restrict__ Xt,
                 const int* __restrict__ cols, const int* __restrict__ rows,
                 float* __restrict__ out) {
    const int tile_t = blockIdx.x;       // 0..15
    const int blk    = blockIdx.y;       // 0..127
    const int b      = blockIdx.z;       // 0..3
    const int tid  = threadIdx.x;
    const int wave = tid >> 5;
    const int lane = tid & 31;
    const int mt = wave & 3;             // M-tile (0..3)
    const int nh = wave >> 2;            // N-half (0..1)
    const int g   = lane >> 4;
    const int l16 = lane & 15;

    const int col = cols[blk];
    const int row = rows[blk];
    const int tstart = tile_t * NT;

    // A fragments: lane-contiguous 32B reads from pre-swizzled Wf
    const _Float16* aptr = Wf + (size_t)(blk * 4 + mt) * 16384 + (size_t)lane * 16;

    const _Float16* xb = Xt + (size_t)b * XT_ROWS * C_;
    const int cbase0 = col * BS + g * 16;       // + (kc&1)*32 per chunk

    v8f acc[4];
    #pragma unroll
    for (int ns = 0; ns < 4; ++ns) acc[ns] = (v8f){0,0,0,0,0,0,0,0};

    #pragma unroll 2
    for (int kc = 0; kc < 32; ++kc) {
        v16h a = *(const v16h*)(aptr + (size_t)kc * 512);
        const int k    = kc >> 1;
        const int cofs = cbase0 + (kc & 1) * 32;
        // time row for this lane's column n = tstart + nh*64 + ns*16 + l16
        const _Float16* bp = xb
            + (size_t)(TP + tstart - (KS - 1) + k + nh * 64 + l16) * C_ + cofs;
        #pragma unroll
        for (int ns = 0; ns < 4; ++ns) {
            v16h bf = *(const v16h*)(bp + (size_t)(ns * 16) * C_);
            acc[ns] = __builtin_amdgcn_wmma_f32_16x16x32_f16(
                false, a, false, bf, (short)0, acc[ns], false, false);
        }
    }

    // Scatter-add: C/D layout: lane l16 = column n, vgpr v -> m = v + 8*g.
    float* obase = out + ((size_t)b * C_ + (size_t)row * BS + mt * 16) * T_;
    #pragma unroll
    for (int ns = 0; ns < 4; ++ns) {
        const int n = tstart + nh * 64 + ns * 16 + l16;
        #pragma unroll
        for (int v = 0; v < 8; ++v) {
            const int m = v + 8 * g;
            atomicAdd(obase + (size_t)m * T_ + n, acc[ns][v]);
        }
    }
}

// ---------------- launcher ----------------
extern "C" void kernel_launch(void* const* d_in, const int* in_sizes, int n_in,
                              void* d_out, int out_size, void* d_ws, size_t ws_size,
                              hipStream_t stream) {
    const float* x  = (const float*)d_in[0];
    const float* bv = (const float*)d_in[1];
    const int* cols = (const int*)d_in[2];
    const int* rows = (const int*)d_in[3];
    float* out = (float*)d_out;

    _Float16* Wf = (_Float16*)d_ws;
    _Float16* Xt = (_Float16*)((char*)d_ws + WF_BYTES);

    zero_f32<<<4096, 256, 0, stream>>>(out, (size_t)out_size);
    prep_w<<<(int)(WF_ELEMS / 256), 256, 0, stream>>>(bv, Wf);
    prep_x<<<dim3(T_ / 32, C_ / 32, B_), dim3(32, 8), 0, stream>>>(x, Xt);
    zero_xpad<<<(B_ * TP * C_) / 256, 256, 0, stream>>>(Xt);
    bsconv_wmma<<<dim3(T_ / NT, NBLK, B_), 256, 0, stream>>>(Wf, Xt, cols, rows, out);
}
